// MultiHeadAttn_9972914061978
// MI455X (gfx1250) — matrix operand order repaired and checked by reading, compiled, not run on hardware
//
#include <hip/hip_runtime.h>
#include <hip/hip_bf16.h>
#include <stdint.h>

// ---------------------------------------------------------------------------
// Problem constants (from reference)
// ---------------------------------------------------------------------------
#define N_HEAD  16
#define D_MODEL 1024
#define D_HEAD  64
#define BSZ     4
#define SEQ     2048
#define NROWS   (BSZ * SEQ)          // 8192
#define SCALE   0.125f               // 1/sqrt(64)
#define LN_EPS  1e-5f
#define NEGINF  -1e10f

typedef __attribute__((ext_vector_type(16))) __bf16 v16bf;
typedef __attribute__((ext_vector_type(8)))  float  v8f;

// ---------------------------------------------------------------------------
// CDNA5 async global->LDS staging (CDNA5 ISA).
// Each lane copies 16B; tracked with ASYNCcnt; loads complete in order, so
// s_wait_asynccnt N leaves only the newest N instructions outstanding.
// ---------------------------------------------------------------------------
__device__ __forceinline__ uint32_t lds_off_of(const void* p) {
  // Generic shared pointer: addr[31:0] is the LDS offset (ISA §10.2 aperture).
  return (uint32_t)(uintptr_t)p;
}
__device__ __forceinline__ void async_ld_b128(uint32_t ldsoff, const void* g) {
  asm volatile("global_load_async_to_lds_b128 %0, %1, off"
               :: "v"(ldsoff), "v"(g) : "memory");
}
__device__ __forceinline__ void wait_async0() {
  asm volatile("s_wait_asynccnt 0x0" ::: "memory");
}
__device__ __forceinline__ void wait_async2() {
  asm volatile("s_wait_asynccnt 0x2" ::: "memory");
}
__device__ __forceinline__ void wait_async4() {
  asm volatile("s_wait_asynccnt 0x4" ::: "memory");
}

// ---------------------------------------------------------------------------
// WMMA fragment loaders (CDNA5 wave32 layouts, CDNA5 ISA §7.12.2)
// ---------------------------------------------------------------------------

// A fragment: 16x32 bf16 tile, row-major, row stride `ld`.
// lane 0-15: row M=lane, K={0..7,16..23}; lane 16-31: K={8..15,24..31}.
// Per lane: 2 contiguous 16B reads -> ds_load_b128 pairs.
__device__ __forceinline__ v16bf frag_a(const __bf16* p, int ld) {
  const int lane = threadIdx.x & 31;
  const int hf   = (lane >> 4) & 1;
  const __bf16* row = p + (size_t)(lane & 15) * ld;
  v16bf f;
#pragma unroll
  for (int i = 0; i < 4; ++i) {
    const int k = hf * 8 + 2 * i;
    f[2 * i]         = row[k];
    f[2 * i + 1]     = row[k + 1];
    f[8 + 2 * i]     = row[16 + k];
    f[8 + 2 * i + 1] = row[16 + k + 1];
  }
  return f;
}

// B fragment from a transposed store: memory holds B^T as N rows x K cols,
// row stride `ld`; element (k,n) = p[n*ld + k]. lanes 0-15 hold K={0..7,16..23}
// for column n=lane, lanes 16-31 hold K={8..15,24..31}.
__device__ __forceinline__ v16bf frag_bt(const __bf16* p, int ld) {
  const int lane = threadIdx.x & 31;
  const int hf   = (lane >> 4) & 1;
  const __bf16* row = p + (size_t)(lane & 15) * ld + hf * 8;
  v16bf f;
#pragma unroll
  for (int i = 0; i < 8; ++i) { f[i] = row[i]; f[8 + i] = row[16 + i]; }
  return f;
}

__device__ __forceinline__ v8f wmma_bf16(v16bf a, v16bf b, v8f c) {
  return __builtin_amdgcn_wmma_f32_16x16x32_bf16(
      false, a, false, b, (short)0, c, false, false);
}

// ---------------------------------------------------------------------------
// Kernel 1: LayerNorm. One block per row. Writes qn (f32, residual base) to
// d_out and a bf16 copy to workspace for the WMMA GEMMs.
// ---------------------------------------------------------------------------
__global__ __launch_bounds__(256) void ln_kernel(
    const float* __restrict__ q, const float* __restrict__ gamma,
    const float* __restrict__ beta, float* __restrict__ outf,
    __bf16* __restrict__ outb) {
  const int row = blockIdx.x;
  const int tid = threadIdx.x;
  const float* x = q + (size_t)row * D_MODEL;

  float v[4], s = 0.f, s2 = 0.f;
#pragma unroll
  for (int i = 0; i < 4; ++i) {
    v[i] = x[tid * 4 + i];
    s  += v[i];
    s2 += v[i] * v[i];
  }
  __shared__ float rs[256], rq[256];
  rs[tid] = s;
  rq[tid] = s2;
  __syncthreads();
  for (int off = 128; off > 0; off >>= 1) {
    if (tid < off) {
      rs[tid] += rs[tid + off];
      rq[tid] += rq[tid + off];
    }
    __syncthreads();
  }
  const float mu   = rs[0] * (1.f / D_MODEL);
  const float var  = rq[0] * (1.f / D_MODEL) - mu * mu;
  const float rstd = rsqrtf(var + LN_EPS);
#pragma unroll
  for (int i = 0; i < 4; ++i) {
    const int c = tid * 4 + i;
    const float y = (v[i] - mu) * rstd * gamma[c] + beta[c];
    outf[(size_t)row * D_MODEL + c] = y;
    outb[(size_t)row * D_MODEL + c] = (__bf16)y;
  }
}

// ---------------------------------------------------------------------------
// Kernel 1b: convert + transpose weights: W (f32, KxN) -> Wt (bf16, NxK).
// Gives contiguous async b128 staging and b128 fragment reads in the GEMMs.
// ---------------------------------------------------------------------------
__global__ __launch_bounds__(256) void wt_kernel(
    const float* __restrict__ W0, const float* __restrict__ W1,
    const float* __restrict__ W2, const float* __restrict__ W3,
    __bf16* __restrict__ T0, __bf16* __restrict__ T1,
    __bf16* __restrict__ T2, __bf16* __restrict__ T3) {
  const float* src;
  __bf16* dst;
  if      (blockIdx.z == 0) { src = W0; dst = T0; }
  else if (blockIdx.z == 1) { src = W1; dst = T1; }
  else if (blockIdx.z == 2) { src = W2; dst = T2; }
  else                      { src = W3; dst = T3; }
  const int r0 = blockIdx.y * 64, c0 = blockIdx.x * 64;
  const int tid = threadIdx.x;
  __shared__ __bf16 Ts[64][65];
#pragma unroll
  for (int i = 0; i < 16; ++i) {
    const int idx = tid + i * 256;
    const int r = idx >> 6, c = idx & 63;
    Ts[r][c] = (__bf16)src[(size_t)(r0 + r) * D_MODEL + c0 + c];
  }
  __syncthreads();
#pragma unroll
  for (int i = 0; i < 16; ++i) {
    const int idx = tid + i * 256;
    const int r = idx >> 6, c = idx & 63;
    dst[(size_t)(c0 + r) * D_MODEL + r0 + c] = Ts[c][r];
  }
}

// ---------------------------------------------------------------------------
// Kernel 2/4: tiled WMMA GEMM with async double-buffered staging.
//   C[M,N] (+)= out_scale * (A_bf16[M,K] @ Bt_bf16[N,K]^T)
// Block tile 128x128, 8 waves each owning a 16x128 strip (8 accumulators).
// Per K-step each wave issues 4 async b128 copies (2 A + 2 Bt); LDS 32 KB.
// ---------------------------------------------------------------------------
template <bool ADD_F32>
__global__ __launch_bounds__(256) void gemm_kernel(
    const __bf16* __restrict__ A, const __bf16* __restrict__ Bt,
    void* __restrict__ C, int M, int N, int K, float out_scale) {
  constexpr int BM = 128, BN = 128, BK = 32;
  __shared__ __bf16 As[2][BM * BK];   // 2 x 8 KB
  __shared__ __bf16 Bs[2][BN * BK];   // 2 x 8 KB (Bt tile: BN rows x BK cols)

  const int tid  = threadIdx.x;
  const int wave = tid >> 5;           // 16-row strip
  const int m0   = blockIdx.y * BM;
  const int n0   = blockIdx.x * BN;

  const uint32_t as_base = lds_off_of(&As[0][0]);
  const uint32_t bs_base = lds_off_of(&Bs[0][0]);

  // Stage one K-step: A 128x32 and Bt 128x32, 512 16B-chunks each,
  // 2 chunks/thread each -> 4 async instructions per wave.
  auto issue_tile = [&](int buf, int k0) {
#pragma unroll
    for (int i = 0; i < 2; ++i) {
      const int c = tid + i * 256;
      const int r = c >> 2, cc = c & 3;
      async_ld_b128(as_base + buf * (BM * BK * 2) + c * 16,
                    A + (size_t)(m0 + r) * K + k0 + cc * 8);
      async_ld_b128(bs_base + buf * (BN * BK * 2) + c * 16,
                    Bt + (size_t)(n0 + r) * K + k0 + cc * 8);
    }
  };

  v8f acc[8];
#pragma unroll
  for (int t = 0; t < 8; ++t) acc[t] = (v8f){0, 0, 0, 0, 0, 0, 0, 0};

  issue_tile(0, 0);
  const int NIT = K / BK;
  for (int it = 0; it < NIT; ++it) {
    const int cur = it & 1;
    if (it + 1 < NIT) {
      issue_tile(cur ^ 1, (it + 1) * BK);
      wait_async4();          // current tile (older 4) complete
    } else {
      wait_async0();
    }
    __syncthreads();          // all waves' current tile staged

    const v16bf a = frag_a(&As[cur][wave * 16 * BK], BK);
#pragma unroll
    for (int t = 0; t < 8; ++t) {
      const v16bf b = frag_bt(&Bs[cur][(t * 16) * BK], BK);
      acc[t] = wmma_bf16(a, b, acc[t]);
    }
    __syncthreads();          // done reading buf[cur] before it is re-staged
  }

  const int lane = tid & 31, hf = lane >> 4, ln = lane & 15;
#pragma unroll
  for (int t = 0; t < 8; ++t) {
#pragma unroll
    for (int r = 0; r < 8; ++r) {
      const int gm = m0 + wave * 16 + hf * 8 + r;
      const int gn = n0 + t * 16 + ln;
      const float y = acc[t][r] * out_scale;
      if (ADD_F32)
        ((float*)C)[(size_t)gm * N + gn] += y;
      else
        ((__bf16*)C)[(size_t)gm * N + gn] = (__bf16)y;
    }
  }
}

// ---------------------------------------------------------------------------
// Kernel 3: flash-style attention per (batch, head).
// Block = 128 query rows (8 waves x 16 rows), key blocks of 64.
// Q + K tiles staged with async b128 (K double-buffered); V staged via
// coalesced 16B global loads and written *transposed* into LDS so the P*V
// B-fragments are contiguous b128 reads. Online softmax row stats live on
// the C-fragment vgpr/half axis, reduced with 16-lane shfl_xor.
// Q is pre-scaled by 1/sqrt(d) at the hq GEMM epilogue, so no per-score mul.
// LDS: Qs 16K + Ks[2] 16K + VtS[2] 16K + Ps 16K = 64 KB.
// ---------------------------------------------------------------------------
__global__ __launch_bounds__(256) void attn_kernel(
    const __bf16* __restrict__ hq, const __bf16* __restrict__ hk,
    const __bf16* __restrict__ hv, const unsigned char* __restrict__ mask,
    __bf16* __restrict__ vec) {
  constexpr int LD = N_HEAD * D_HEAD;  // 1024
  const int bh = blockIdx.y;
  const int b  = bh >> 4;
  const int h  = bh & 15;
  const int i0 = blockIdx.x * 128;

  const int tid  = threadIdx.x;
  const int wave = tid >> 5;
  const int lane = tid & 31;
  const int hf   = lane >> 4;
  const int ln   = lane & 15;

  __shared__ __bf16 Qs[128 * 64];      // 16 KB
  __shared__ __bf16 Ks[2][64 * 64];    // 16 KB
  __shared__ __bf16 VtS[2][64 * 64];   // 16 KB (transposed: [d][j])
  __shared__ __bf16 Ps[128 * 64];      // 16 KB

  const __bf16* qb = hq + ((size_t)(b * SEQ + i0)) * LD + h * D_HEAD;
  const __bf16* kb = hk + ((size_t)b * SEQ) * LD + h * D_HEAD;
  const __bf16* vb = hv + ((size_t)b * SEQ) * LD + h * D_HEAD;

  // Async-stage Q tile: 128x64 = 1024 chunks, 4/thread (4 instr/wave).
  {
    const uint32_t q_base = lds_off_of(&Qs[0]);
#pragma unroll
    for (int i = 0; i < 4; ++i) {
      const int c = tid + i * 256;
      const int r = c >> 3, cc = c & 7;
      async_ld_b128(q_base + c * 16, qb + (size_t)r * LD + cc * 8);
    }
  }
  // Async-stage a K tile: 64x64 = 512 chunks, 2/thread (2 instr/wave).
  const uint32_t k_base = lds_off_of(&Ks[0][0]);
  auto issue_k = [&](int buf, int j0) {
#pragma unroll
    for (int i = 0; i < 2; ++i) {
      const int c = tid + i * 256;
      const int r = c >> 3, cc = c & 7;
      async_ld_b128(k_base + buf * 8192 + c * 16,
                    kb + (size_t)(j0 + r) * LD + cc * 8);
    }
  };
  issue_k(0, 0);

  float m_i[8], l_i[8];
  v8f o[4];
#pragma unroll
  for (int r = 0; r < 8; ++r) { m_i[r] = -1e30f; l_i[r] = 0.f; }
#pragma unroll
  for (int t = 0; t < 4; ++t) o[t] = (v8f){0, 0, 0, 0, 0, 0, 0, 0};

  const int NJ = SEQ / 64;
  for (int it = 0; it < NJ; ++it) {
    const int j0  = it * 64;
    const int cur = it & 1;

    if (it + 1 < NJ) issue_k(cur ^ 1, j0 + 64);

    // V tile for this iteration: coalesced 16B register loads (latency
    // hidden behind the async wait + S-phase WMMAs below).
    uint4 vr[2];
#pragma unroll
    for (int i = 0; i < 2; ++i) {
      const int c = tid + i * 256;
      const int r = c >> 3, cc = c & 7;
      vr[i] = *reinterpret_cast<const uint4*>(vb + (size_t)(j0 + r) * LD + cc * 8);
    }

    if (it + 1 < NJ) wait_async2(); else wait_async0();
    __syncthreads();  // Q (iter 0) and K[cur] staged for all waves

    // ---- S = Q*K^T : contract over d (2 WMMA steps) ----
    v8f s[4];
#pragma unroll
    for (int t = 0; t < 4; ++t) s[t] = (v8f){0, 0, 0, 0, 0, 0, 0, 0};
#pragma unroll
    for (int k0 = 0; k0 < D_HEAD; k0 += 32) {
      const v16bf a = frag_a(Qs + wave * 16 * 64 + k0, 64);
#pragma unroll
      for (int t = 0; t < 4; ++t) {
        const v16bf bt = frag_bt(&Ks[cur][(t * 16) * 64 + k0], 64);
        s[t] = wmma_bf16(a, bt, s[t]);
      }
    }

    // ---- mask + online softmax (Q already carries 1/sqrt(d)) ----
    float pm[8];
#pragma unroll
    for (int r = 0; r < 8; ++r) pm[r] = -1e30f;
#pragma unroll
    for (int t = 0; t < 4; ++t) {
      const int jg = j0 + t * 16 + ln;
#pragma unroll
      for (int r = 0; r < 8; ++r) {
        const int ig = i0 + wave * 16 + hf * 8 + r;
        float xv = s[t][r];
        if (mask[(size_t)ig * SEQ + jg]) xv = NEGINF;
        s[t][r] = xv;
        pm[r] = fmaxf(pm[r], xv);
      }
    }
#pragma unroll
    for (int r = 0; r < 8; ++r) {
#pragma unroll
      for (int off = 1; off < 16; off <<= 1)
        pm[r] = fmaxf(pm[r], __shfl_xor(pm[r], off, 32));
    }
    float pl[8];
#pragma unroll
    for (int r = 0; r < 8; ++r) {
      const float mnew = fmaxf(m_i[r], pm[r]);
      const float fac  = __expf(m_i[r] - mnew);
      m_i[r] = mnew;
      l_i[r] *= fac;
      pl[r] = 0.f;
#pragma unroll
      for (int t = 0; t < 4; ++t) o[t][r] *= fac;
    }
#pragma unroll
    for (int t = 0; t < 4; ++t) {
#pragma unroll
      for (int r = 0; r < 8; ++r) {
        const float p = __expf(s[t][r] - m_i[r]);
        pl[r] += p;
        Ps[wave * 1024 + (hf * 8 + r) * 64 + t * 16 + ln] = (__bf16)p;
      }
    }
#pragma unroll
    for (int r = 0; r < 8; ++r) {
#pragma unroll
      for (int off = 1; off < 16; off <<= 1)
        pl[r] += __shfl_xor(pl[r], off, 32);
      l_i[r] += pl[r];
    }

    // ---- write V tile transposed into LDS: VtS[d][j] ----
#pragma unroll
    for (int i = 0; i < 2; ++i) {
      const int c = tid + i * 256;
      const int r = c >> 3, cc = (c & 7) * 8;
      const __bf16* pv = reinterpret_cast<const __bf16*>(&vr[i]);
#pragma unroll
      for (int e = 0; e < 8; ++e)
        VtS[cur][(cc + e) * 64 + r] = pv[e];
    }
    __syncthreads();  // VtS[cur] + Ps visible to all waves

    // ---- O += P*V : contract over j (2 WMMA steps), contiguous b128 ----
#pragma unroll
    for (int k0 = 0; k0 < 64; k0 += 32) {
      const v16bf a = frag_a(Ps + wave * 1024 + k0, 64);
#pragma unroll
      for (int t = 0; t < 4; ++t) {
        const v16bf bv = frag_bt(&VtS[cur][(t * 16) * 64 + k0], 64);
        o[t] = wmma_bf16(a, bv, o[t]);
      }
    }
    // No trailing barrier needed: next iteration re-stages Ks[cur^1]
    // (read finished before the mid barrier) and VtS[cur^1] (read two
    // iterations ago, separated by barriers).
  }

  // ---- normalize + store vec (bf16) ----
#pragma unroll
  for (int t = 0; t < 4; ++t) {
#pragma unroll
    for (int r = 0; r < 8; ++r) {
      const int ig = i0 + wave * 16 + hf * 8 + r;
      const int cg = h * D_HEAD + t * 16 + ln;
      vec[((size_t)b * SEQ + ig) * LD + cg] = (__bf16)(o[t][r] / l_i[r]);
    }
  }
}

// ---------------------------------------------------------------------------
// Launch: LN -> Wt convert -> 3x QKV GEMM -> attention -> Wo GEMM (+residual)
// ---------------------------------------------------------------------------
extern "C" void kernel_launch(void* const* d_in, const int* in_sizes, int n_in,
                              void* d_out, int out_size, void* d_ws,
                              size_t ws_size, hipStream_t stream) {
  const float* q     = (const float*)d_in[0];
  const unsigned char* mask = (const unsigned char*)d_in[1];  // jax bool = 1B
  const float* Wq    = (const float*)d_in[2];
  const float* Wk    = (const float*)d_in[3];
  const float* Wv    = (const float*)d_in[4];
  const float* Wo    = (const float*)d_in[5];
  const float* gamma = (const float*)d_in[6];
  const float* beta  = (const float*)d_in[7];
  float* out = (float*)d_out;

  const size_t SLAB = (size_t)NROWS * D_MODEL * sizeof(__bf16);   // 16 MB
  const size_t WSLB = (size_t)D_MODEL * D_MODEL * sizeof(__bf16); //  2 MB
  char* ws = (char*)d_ws;
  __bf16* qn_bf = (__bf16*)(ws);
  __bf16* hq    = (__bf16*)(ws + 1 * SLAB);
  __bf16* hk    = (__bf16*)(ws + 2 * SLAB);
  __bf16* hv    = (__bf16*)(ws + 3 * SLAB);
  __bf16* vec   = (__bf16*)(ws + 4 * SLAB);
  __bf16* WqT   = (__bf16*)(ws + 5 * SLAB);
  __bf16* WkT   = (__bf16*)(ws + 5 * SLAB + 1 * WSLB);
  __bf16* WvT   = (__bf16*)(ws + 5 * SLAB + 2 * WSLB);
  __bf16* WoT   = (__bf16*)(ws + 5 * SLAB + 3 * WSLB);

  ln_kernel<<<NROWS, 256, 0, stream>>>(q, gamma, beta, out, qn_bf);
  wt_kernel<<<dim3(16, 16, 4), 256, 0, stream>>>(Wq, Wk, Wv, Wo,
                                                 WqT, WkT, WvT, WoT);

  dim3 gg(D_MODEL / 128, NROWS / 128);  // (8, 64)
  // Fold 1/sqrt(d_head) into hq so attention needs no per-score scaling.
  gemm_kernel<false><<<gg, 256, 0, stream>>>(qn_bf, WqT, hq, NROWS, D_MODEL,
                                             D_MODEL, SCALE);
  gemm_kernel<false><<<gg, 256, 0, stream>>>(qn_bf, WkT, hk, NROWS, D_MODEL,
                                             D_MODEL, 1.0f);
  gemm_kernel<false><<<gg, 256, 0, stream>>>(qn_bf, WvT, hv, NROWS, D_MODEL,
                                             D_MODEL, 1.0f);

  dim3 ga(SEQ / 128, BSZ * N_HEAD);    // (16, 64)
  attn_kernel<<<ga, 256, 0, stream>>>(hq, hk, hv, mask, vec);

  gemm_kernel<true><<<gg, 256, 0, stream>>>(vec, WoT, out, NROWS, D_MODEL,
                                            D_MODEL, 1.0f);
}
